// PSPGO_17892833755265
// MI455X (gfx1250) — matrix operand order repaired and checked by reading, not compiled, not run on hardware
//
#include <hip/hip_runtime.h>
#include <math.h>

// ---------------------------------------------------------------------------
// Types for WMMA (gfx1250, wave32)
// ---------------------------------------------------------------------------
typedef __attribute__((ext_vector_type(16))) __bf16 v16bf;
typedef __attribute__((ext_vector_type(8)))  float  v8f;

#define H_DIM 128
#define OUT_DIM 256
#define LN_EPS 1e-5f
#define PF_EDGES 4096   // prefetch distance (edges) for gather streams

// bf16 round-to-nearest-even from f32 (bit pattern in unsigned short)
__device__ __forceinline__ unsigned short f2bf(float f) {
  unsigned u = __float_as_uint(f);
  unsigned r = u + 0x7FFFu + ((u >> 16) & 1u);
  return (unsigned short)(r >> 16);
}

// monotone float <-> uint key for atomicMax-based segment_max
__device__ __forceinline__ unsigned fkey(float f) {
  unsigned u = __float_as_uint(f);
  return (u & 0x80000000u) ? ~u : (u | 0x80000000u);
}
__device__ __forceinline__ float unkey(unsigned k) {
  return (k & 0x80000000u) ? __uint_as_float(k ^ 0x80000000u)
                           : __uint_as_float(~k);
}

// ---------------------------------------------------------------------------
// Weight convert + transpose: W[K][Ncol] f32 -> Wt[Ncol][K] bf16
// ---------------------------------------------------------------------------
__global__ void k_convT(const float* __restrict__ W, unsigned short* __restrict__ Wt,
                        int K, int Ncol) {
  int id = blockIdx.x * blockDim.x + threadIdx.x;
  if (id >= K * Ncol) return;
  int k = id / Ncol, n = id % Ncol;
  Wt[(size_t)n * K + k] = f2bf(W[(size_t)k * Ncol + n]);
}

__global__ void k_fill_u32(unsigned* __restrict__ p, unsigned v, int n) {
  int id = blockIdx.x * blockDim.x + threadIdx.x;
  if (id < n) p[id] = v;
}

// ---------------------------------------------------------------------------
// EmbeddingBag(sum, per_sample_weights) + bias + ReLU. block = 128 (one node)
// ---------------------------------------------------------------------------
__global__ void k_embed_bag(const int* __restrict__ idx, const int* __restrict__ off,
                            const float* __restrict__ w, const float* __restrict__ EW,
                            const float* __restrict__ eb,
                            float* __restrict__ h, unsigned short* __restrict__ hb) {
  int n = blockIdx.x, d = threadIdx.x;
  int s = off[n], e = off[n + 1];
  float acc = 0.f;
  for (int j = s; j < e; ++j)
    acc = fmaf(EW[(size_t)idx[j] * H_DIM + d], w[j], acc);
  acc = fmaxf(acc + eb[d], 0.f);
  h[(size_t)n * H_DIM + d] = acc;
  hb[(size_t)n * H_DIM + d] = f2bf(acc);
}

// ---------------------------------------------------------------------------
// WMMA GEMM: C[M][Ncol] = A[M][K](bf16) * Wt[Ncol][K](bf16,transposed) + bias
// grid = (M/16, Ncol/64), block = 128 (4 waves, one 16x16 tile each)
// ---------------------------------------------------------------------------
__global__ void k_gemm_bf16(const unsigned short* __restrict__ A,
                            const unsigned short* __restrict__ Bt,
                            const float* __restrict__ bias,
                            float* __restrict__ C, int K, int Ncol) {
  const int lane  = threadIdx.x & 31;
  const int wave  = threadIdx.x >> 5;
  const int tileM = blockIdx.x;
  const int tileN = blockIdx.y * 4 + wave;
  const int hi    = lane >> 4;        // 0: lanes 0-15, 1: lanes 16-31
  const int l15   = lane & 15;

  // A fragment: row = tileM*16 + l15; K chunks {koff..koff+7, koff+16..koff+23}, koff = 8*hi
  const unsigned short* ap = A  + (size_t)(tileM * 16 + l15) * K + hi * 8;
  // B fragment (from transposed weights): col = tileN*16 + l15; K chunks {koff..koff+7, koff+16..koff+23}, koff = 8*hi
  const unsigned short* bp = Bt + (size_t)(tileN * 16 + l15) * K + hi * 8;

  union U { uint4 q[2]; v16bf v; };
  v8f acc = {};
  for (int k0 = 0; k0 < K; k0 += 32) {
    U a, b;
    a.q[0] = *(const uint4*)(ap + k0);
    a.q[1] = *(const uint4*)(ap + k0 + 16);
    b.q[0] = *(const uint4*)(bp + k0);
    b.q[1] = *(const uint4*)(bp + k0 + 16);
    acc = __builtin_amdgcn_wmma_f32_16x16x32_bf16(false, a.v, false, b.v,
                                                  (short)0, acc, false, false);
  }
  const int col = tileN * 16 + l15;
  const float bv = bias ? bias[col] : 0.f;
  const int rowbase = tileM * 16 + hi * 8;   // C: VGPR r -> row r (+8 for upper half)
#pragma unroll
  for (int r = 0; r < 8; ++r)
    C[(size_t)(rowbase + r) * Ncol + col] = acc[r] + bv;
}

// ---------------------------------------------------------------------------
// LayerNorm + ReLU over dim=128. block = 128 (one row)
// ---------------------------------------------------------------------------
__global__ void k_ln_relu(const float* __restrict__ X, const float* __restrict__ g,
                          const float* __restrict__ b,
                          float* __restrict__ h, unsigned short* __restrict__ hb) {
  __shared__ float sh[H_DIM];
  int n = blockIdx.x, d = threadIdx.x;
  float v = X[(size_t)n * H_DIM + d];
  sh[d] = v; __syncthreads();
  for (int o = 64; o > 0; o >>= 1) { if (d < o) sh[d] += sh[d + o]; __syncthreads(); }
  float mu = sh[0] * (1.f / H_DIM); __syncthreads();
  float c = v - mu;
  sh[d] = c * c; __syncthreads();
  for (int o = 64; o > 0; o >>= 1) { if (d < o) sh[d] += sh[d + o]; __syncthreads(); }
  float var = sh[0] * (1.f / H_DIM);
  float out = fmaxf(c * rsqrtf(var + LN_EPS) * g[d] + b[d], 0.f);
  h[(size_t)n * H_DIM + d] = out;
  hb[(size_t)n * H_DIM + d] = f2bf(out);
}

// ---------------------------------------------------------------------------
// Edge score: s[e] = sum_d attn[d]*leaky(f[src]+f[dst]); segment_max via keyed atomicMax
// one wave per edge, 4 dims/lane (float4). block = 256 -> 8 edges.
// Prefetches feature rows for edges PF_EDGES ahead into L2 (192MB: whole f fits).
// ---------------------------------------------------------------------------
__global__ void k_edge_score(const float* __restrict__ f, const int* __restrict__ src,
                             const int* __restrict__ dst, const float* __restrict__ attn,
                             float* __restrict__ s, unsigned* __restrict__ nodemax, int E) {
  int e = blockIdx.x * 8 + (threadIdx.x >> 5);
  int lane = threadIdx.x & 31;
  if (e >= E) return;
  if (lane == 0 && e + PF_EDGES < E) {
    __builtin_prefetch(f + (size_t)src[e + PF_EDGES] * H_DIM, 0, 1);
    __builtin_prefetch(f + (size_t)dst[e + PF_EDGES] * H_DIM, 0, 1);
  }
  int sn = src[e], dn = dst[e];
  float4 fs = *(const float4*)(f + (size_t)sn * H_DIM + lane * 4);
  float4 fd = *(const float4*)(f + (size_t)dn * H_DIM + lane * 4);
  float4 at = *(const float4*)(attn + lane * 4);
  float t0 = fs.x + fd.x; t0 = t0 > 0.f ? t0 : 0.2f * t0;
  float t1 = fs.y + fd.y; t1 = t1 > 0.f ? t1 : 0.2f * t1;
  float t2 = fs.z + fd.z; t2 = t2 > 0.f ? t2 : 0.2f * t2;
  float t3 = fs.w + fd.w; t3 = t3 > 0.f ? t3 : 0.2f * t3;
  float v = t0 * at.x + t1 * at.y + t2 * at.z + t3 * at.w;
  for (int o = 16; o > 0; o >>= 1) v += __shfl_xor(v, o, 32);
  if (lane == 0) {
    s[e] = v;
    atomicMax(&nodemax[dn], fkey(v));
  }
}

// ex = exp(s - max[dst]); den[dst] += ex. one thread per edge
__global__ void k_edge_exp(float* __restrict__ s, const int* __restrict__ dst,
                           const unsigned* __restrict__ nodemax,
                           float* __restrict__ den, int E) {
  int e = blockIdx.x * blockDim.x + threadIdx.x;
  if (e >= E) return;
  int dn = dst[e];
  float ex = __expf(s[e] - unkey(nodemax[dn]));
  s[e] = ex;
  atomicAdd(&den[dn], ex);
}

// a = ex/den[dst]; rst[dst] += f[src]*a. one wave per edge
__global__ void k_edge_apply(float* __restrict__ s, const float* __restrict__ f,
                             const int* __restrict__ src, const int* __restrict__ dst,
                             const float* __restrict__ den, float* __restrict__ rst, int E) {
  int e = blockIdx.x * 8 + (threadIdx.x >> 5);
  int lane = threadIdx.x & 31;
  if (e >= E) return;
  if (lane == 0 && e + PF_EDGES < E)
    __builtin_prefetch(f + (size_t)src[e + PF_EDGES] * H_DIM, 0, 1);
  int sn = src[e], dn = dst[e];
  float a = s[e] / den[dn];
  float4 fv = *(const float4*)(f + (size_t)sn * H_DIM + lane * 4);
  float* rp = rst + (size_t)dn * H_DIM + lane * 4;
  atomicAdd(rp + 0, fv.x * a);
  atomicAdd(rp + 1, fv.y * a);
  atomicAdd(rp + 2, fv.z * a);
  atomicAdd(rp + 3, fv.w * a);
  if (lane == 0) s[e] = a;   // persist attention coeff for label prop
}

// h = elu(rst_p + rst_s + 2*h)  (== elu(h_p + h_s), residual folded in)
__global__ void k_combine(const float* __restrict__ rp, const float* __restrict__ rs,
                          float* __restrict__ h, unsigned short* __restrict__ hb, int n) {
  int i = blockIdx.x * blockDim.x + threadIdx.x;
  if (i >= n) return;
  float v = rp[i] + rs[i] + 2.f * h[i];
  v = v > 0.f ? v : (__expf(v) - 1.f);
  h[i] = v;
  hb[i] = f2bf(v);
}

// yacc[dst] += y[src]*a. one wave per edge, 8 dims/lane over 256
__global__ void k_label(const float* __restrict__ y, const int* __restrict__ src,
                        const int* __restrict__ dst, const float* __restrict__ a,
                        float* __restrict__ yacc, int E) {
  int e = blockIdx.x * 8 + (threadIdx.x >> 5);
  int lane = threadIdx.x & 31;
  if (e >= E) return;
  if (lane == 0 && e + PF_EDGES < E) {
    const float* yp = y + (size_t)src[e + PF_EDGES] * OUT_DIM;
    __builtin_prefetch(yp, 0, 1);
    __builtin_prefetch(yp + 128, 0, 1);   // row is 1KB; touch second half too
  }
  int sn = src[e], dn = dst[e];
  float av = a[e];
  const float* yr = y + (size_t)sn * OUT_DIM + lane * 8;
  float* yo = yacc + (size_t)dn * OUT_DIM + lane * 8;
  float4 y0 = *(const float4*)(yr);
  float4 y1 = *(const float4*)(yr + 4);
  atomicAdd(yo + 0, y0.x * av); atomicAdd(yo + 1, y0.y * av);
  atomicAdd(yo + 2, y0.z * av); atomicAdd(yo + 3, y0.w * av);
  atomicAdd(yo + 4, y1.x * av); atomicAdd(yo + 5, y1.y * av);
  atomicAdd(yo + 6, y1.z * av); atomicAdd(yo + 7, y1.w * av);
}

// yh = flag ? y_in : yacc / max(||yacc||, 1e-12). block = 256 (one node)
__global__ void k_ynorm(const float* __restrict__ yacc, const float* __restrict__ yin,
                        const unsigned char* __restrict__ flag, float* __restrict__ yout) {
  __shared__ float sh[OUT_DIM];
  int n = blockIdx.x, d = threadIdx.x;
  float v = yacc[(size_t)n * OUT_DIM + d];
  sh[d] = v * v; __syncthreads();
  for (int o = 128; o > 0; o >>= 1) { if (d < o) sh[d] += sh[d + o]; __syncthreads(); }
  float inv = 1.f / fmaxf(sqrtf(sh[0]), 1e-12f);
  float out = flag[n] ? yin[(size_t)n * OUT_DIM + d] : v * inv;
  yout[(size_t)n * OUT_DIM + d] = out;
}

// ---------------------------------------------------------------------------
// Launch
// ---------------------------------------------------------------------------
extern "C" void kernel_launch(void* const* d_in, const int* in_sizes, int n_in,
                              void* d_out, int out_size, void* d_ws, size_t ws_size,
                              hipStream_t stream) {
  const int N = in_sizes[1] - 1;          // 50000
  const int E = in_sizes[5];              // 800000
  const int H = H_DIM, OD = OUT_DIM;

  const int*   emb_idx = (const int*)  d_in[0];
  const int*   emb_off = (const int*)  d_in[1];
  const float* emb_w   = (const float*)d_in[2];
  const float* y0      = (const float*)d_in[3];
  const unsigned char* flag = (const unsigned char*)d_in[4];
  const int* src_p1 = (const int*)d_in[5];  const int* dst_p1 = (const int*)d_in[6];
  const int* src_s1 = (const int*)d_in[7];  const int* dst_s1 = (const int*)d_in[8];
  const int* src_p2 = (const int*)d_in[9];  const int* dst_p2 = (const int*)d_in[10];
  const int* src_s2 = (const int*)d_in[11]; const int* dst_s2 = (const int*)d_in[12];
  const float* embed_W = (const float*)d_in[13];
  const float* embed_b = (const float*)d_in[14];
  const float* mlp_W1  = (const float*)d_in[15]; const float* mlp_b1 = (const float*)d_in[16];
  const float* ln1_g   = (const float*)d_in[17]; const float* ln1_b  = (const float*)d_in[18];
  const float* mlp_W2  = (const float*)d_in[19]; const float* mlp_b2 = (const float*)d_in[20];
  const float* ln2_g   = (const float*)d_in[21]; const float* ln2_b  = (const float*)d_in[22];
  const float* Wsrc1   = (const float*)d_in[23]; const float* bsrc1  = (const float*)d_in[24];
  const float* attn1   = (const float*)d_in[25];
  const float* Wsrc2   = (const float*)d_in[26]; const float* bsrc2  = (const float*)d_in[27];
  const float* attn2   = (const float*)d_in[28];
  const float* out_W   = (const float*)d_in[29]; const float* out_b  = (const float*)d_in[30];

  float* out_logits = (float*)d_out;
  float* out_yh     = (float*)d_out + (size_t)N * OD;

  // workspace bump allocator (256B aligned)
  char* wp = (char*)d_ws;
  auto carve = [&](size_t bytes) -> void* {
    void* r = (void*)wp;
    wp += (bytes + 255) & ~(size_t)255;
    return r;
  };
  float*          h       = (float*)         carve((size_t)N * H * 4);
  unsigned short* hb      = (unsigned short*)carve((size_t)N * H * 2);
  float*          f       = (float*)         carve((size_t)N * H * 4);
  float*          rst_p   = (float*)         carve((size_t)N * H * 4);
  float*          rst_s   = (float*)         carve((size_t)N * H * 4);
  float*          sbuf_p  = (float*)         carve((size_t)E * 4);
  float*          sbuf_s  = (float*)         carve((size_t)E * 4);
  unsigned*       nodemax = (unsigned*)      carve((size_t)N * 4);
  float*          nodeden = (float*)         carve((size_t)N * 4);
  float*          yacc    = (float*)         carve((size_t)N * OD * 4);
  float*          y1      = (float*)         carve((size_t)N * OD * 4);
  unsigned short* wt_mlp1 = (unsigned short*)carve((size_t)H * H * 2);
  unsigned short* wt_mlp2 = (unsigned short*)carve((size_t)H * H * 2);
  unsigned short* wt_fc1  = (unsigned short*)carve((size_t)H * H * 2);
  unsigned short* wt_fc2  = (unsigned short*)carve((size_t)H * H * 2);
  unsigned short* wt_out  = (unsigned short*)carve((size_t)H * OD * 2);

  const unsigned NEG_INF_KEY = 0x007FFFFFu;   // fkey(-inf)

  // 1) weights -> bf16, transposed for contiguous B fragments
  k_convT<<<(H * H + 255) / 256, 256, 0, stream>>>(mlp_W1, wt_mlp1, H, H);
  k_convT<<<(H * H + 255) / 256, 256, 0, stream>>>(mlp_W2, wt_mlp2, H, H);
  k_convT<<<(H * H + 255) / 256, 256, 0, stream>>>(Wsrc1, wt_fc1, H, H);
  k_convT<<<(H * H + 255) / 256, 256, 0, stream>>>(Wsrc2, wt_fc2, H, H);
  k_convT<<<(H * OD + 255) / 256, 256, 0, stream>>>(out_W, wt_out, H, OD);

  // 2) EmbeddingBag + ReLU
  k_embed_bag<<<N, H, 0, stream>>>(emb_idx, emb_off, emb_w, embed_W, embed_b, h, hb);

  // 3) MLP: (GEMM -> LN -> ReLU) x 2
  dim3 g128((unsigned)(N / 16), (unsigned)(H / 64));
  k_gemm_bf16<<<g128, 128, 0, stream>>>(hb, wt_mlp1, mlp_b1, f, H, H);
  k_ln_relu<<<N, H, 0, stream>>>(f, ln1_g, ln1_b, h, hb);
  k_gemm_bf16<<<g128, 128, 0, stream>>>(hb, wt_mlp2, mlp_b2, f, H, H);
  k_ln_relu<<<N, H, 0, stream>>>(f, ln2_g, ln2_b, h, hb);

  // 4) two propagation steps
  auto gat_rel = [&](const int* src, const int* dst, const float* attn,
                     float* sbuf, float* rst) {
    hipMemsetAsync(rst, 0, (size_t)N * H * 4, stream);
    k_fill_u32<<<(N + 255) / 256, 256, 0, stream>>>(nodemax, NEG_INF_KEY, N);
    hipMemsetAsync(nodeden, 0, (size_t)N * 4, stream);
    k_edge_score<<<(E + 7) / 8, 256, 0, stream>>>(f, src, dst, attn, sbuf, nodemax, E);
    k_edge_exp<<<(E + 255) / 256, 256, 0, stream>>>(sbuf, dst, nodemax, nodeden, E);
    k_edge_apply<<<(E + 7) / 8, 256, 0, stream>>>(sbuf, f, src, dst, nodeden, rst, E);
  };

  auto prop = [&](const int* sp, const int* dp, const int* ss, const int* ds,
                  const unsigned short* wt, const float* bsrc, const float* attn,
                  const float* yin, float* yout) {
    k_gemm_bf16<<<g128, 128, 0, stream>>>(hb, wt, bsrc, f, H, H);
    gat_rel(sp, dp, attn, sbuf_p, rst_p);
    gat_rel(ss, ds, attn, sbuf_s, rst_s);
    k_combine<<<(N * H + 255) / 256, 256, 0, stream>>>(rst_p, rst_s, h, hb, N * H);
    hipMemsetAsync(yacc, 0, (size_t)N * OD * 4, stream);
    k_label<<<(E + 7) / 8, 256, 0, stream>>>(yin, sp, dp, sbuf_p, yacc, E);
    k_label<<<(E + 7) / 8, 256, 0, stream>>>(yin, ss, ds, sbuf_s, yacc, E);
    k_ynorm<<<N, OD, 0, stream>>>(yacc, yin, flag, yout);
  };

  prop(src_p1, dst_p1, src_s1, dst_s1, wt_fc1, bsrc1, attn1, y0, y1);
  prop(src_p2, dst_p2, src_s2, dst_s2, wt_fc2, bsrc2, attn2, y1, out_yh);

  // 5) final projection 128 -> 256 (WMMA)
  dim3 g256((unsigned)(N / 16), (unsigned)(OD / 64));
  k_gemm_bf16<<<g256, 128, 0, stream>>>(hb, wt_out, out_b, out_logits, H, OD);
}
